// GRUAttentionDecoder_3882650435838
// MI455X (gfx1250) — compile-verified, matrix operand order checked by reading, not run on hardware
//
#include <hip/hip_runtime.h>
#include <hip/hip_bf16.h>
#include <math.h>

// ---- problem dims ----
constexpr int D_IN  = 512;
constexpr int D_H   = 512;
constexpr int D_H2  = 1024;   // 2*H
constexpr int D_CTX = 1024;
constexpr int N_TY  = 128;
constexpr int N_B   = 64;
constexpr int N_TX  = 512;

typedef __attribute__((ext_vector_type(16))) __bf16 v16bf;
typedef __attribute__((ext_vector_type(8)))  __bf16 v8bf;
typedef __attribute__((ext_vector_type(4)))  __bf16 v4bf;
typedef __attribute__((ext_vector_type(8)))  float  v8f;

// Load one 16-lane-row bf16 WMMA fragment: per ISA 16-bit A/B layout,
// lane holds K = {kh..kh+7} U {kh+16..kh+23} of its row.
__device__ __forceinline__ v16bf ldfrag(const __bf16* p) {
    v8bf lo = *(const v8bf*)p;
    v8bf hi = *(const v8bf*)(p + 16);
    v16bf f;
#pragma unroll
    for (int i = 0; i < 8; ++i) { f[i] = lo[i]; f[8 + i] = hi[i]; }
    return f;
}

__device__ __forceinline__ v8f wmma_bf16(v16bf a, v16bf b, v8f c) {
    return __builtin_amdgcn_wmma_f32_16x16x32_bf16(false, a, false, b, (short)0, c,
                                                   false, false);
}

// Accumulate acc[0..3] += A[m0:m0+16, :K] @ Wt[n0:n0+64, :K]^T  (bf16 in, f32 acc)
// A row-major [M,K] (stride K), Wt row-major [N,K] (stride K).
// Software-pipelined: next iteration's fragments are loaded before the WMMAs
// consuming the current ones, so loadcnt waits are partial and loads overlap
// matrix issue.
__device__ __forceinline__ void accum4(v8f acc[4], const __bf16* A, const __bf16* Wt,
                                       int m0, int n0, int K, int lane) {
    const int r  = lane & 15;
    const int kh = (lane >> 4) * 8;
    const __bf16* pa = A  + (size_t)(m0 + r) * K + kh;
    const __bf16* pb = Wt + (size_t)(n0 + r) * K + kh;
    v16bf a  = ldfrag(pa);
    v16bf b0 = ldfrag(pb);
    v16bf b1 = ldfrag(pb + (size_t)16 * K);
    v16bf b2 = ldfrag(pb + (size_t)32 * K);
    v16bf b3 = ldfrag(pb + (size_t)48 * K);
    for (int k = 32; k < K; k += 32) {
        v16bf an  = ldfrag(pa + k);
        v16bf bn0 = ldfrag(pb + k);
        v16bf bn1 = ldfrag(pb + (size_t)16 * K + k);
        v16bf bn2 = ldfrag(pb + (size_t)32 * K + k);
        v16bf bn3 = ldfrag(pb + (size_t)48 * K + k);
        acc[0] = wmma_bf16(a, b0, acc[0]);
        acc[1] = wmma_bf16(a, b1, acc[1]);
        acc[2] = wmma_bf16(a, b2, acc[2]);
        acc[3] = wmma_bf16(a, b3, acc[3]);
        a = an; b0 = bn0; b1 = bn1; b2 = bn2; b3 = bn3;
    }
    acc[0] = wmma_bf16(a, b0, acc[0]);
    acc[1] = wmma_bf16(a, b1, acc[1]);
    acc[2] = wmma_bf16(a, b2, acc[2]);
    acc[3] = wmma_bf16(a, b3, acc[3]);
}

// ---- fp32 -> bf16 convert (vectorized: float4 in, 4x bf16 out) ----
__global__ void k_cvt(const float* __restrict__ s, __bf16* __restrict__ d, int n4) {
    int i = blockIdx.x * 256 + threadIdx.x;
    if (i < n4) {
        float4 v = *(const float4*)(s + (size_t)i * 4);
        v4bf o;
        o[0] = (__bf16)v.x; o[1] = (__bf16)v.y; o[2] = (__bf16)v.z; o[3] = (__bf16)v.w;
        *(v4bf*)(d + (size_t)i * 4) = o;
    }
}

// ---- generic GEMM: out = A[M,K] @ Wt[N,K]^T + bias ; optional f32/bf16 outputs ----
__global__ void k_gemm(const __bf16* __restrict__ A, const __bf16* __restrict__ Wt,
                       const float* __restrict__ bias,
                       float* __restrict__ outF, __bf16* __restrict__ outB,
                       int M, int N, int K) {
    const int lane = threadIdx.x & 31;
    const int id   = blockIdx.x * 8 + (threadIdx.x >> 5);
    const int nT   = N >> 6;
    const int m0   = (id / nT) * 16;
    const int n0   = (id % nT) * 64;
    v8f acc[4] = {};
    accum4(acc, A, Wt, m0, n0, K, lane);
    const int col = lane & 15, rs = (lane >> 4) * 8;
#pragma unroll
    for (int j = 0; j < 4; ++j) {
        int n = n0 + j * 16 + col;
        float bv = bias ? bias[n] : 0.f;
#pragma unroll
        for (int i = 0; i < 8; ++i) {
            int m = m0 + rs + i;
            float v = acc[j][i] + bv;
            size_t idx = (size_t)m * N + n;
            if (outF) outF[idx] = v;
            if (outB) outB[idx] = (__bf16)v;
        }
    }
}

// ---- h state init ----
__global__ void k_init(const float* __restrict__ s, float* __restrict__ hf,
                       __bf16* __restrict__ hb) {
    int i = blockIdx.x * 256 + threadIdx.x;
    float v = s[i];
    hf[i] = v; hb[i] = (__bf16)v;
}

__device__ __forceinline__ float sigmoidf_(float v) { return 1.f / (1.f + __expf(-v)); }

// ---- step1: tmp1 = sigmoid(h @ U^T + x_t); emit hr = h*r1 (bf16) and u1 ----
__global__ void k_step1(const __bf16* __restrict__ h_bf, const __bf16* __restrict__ U_bf,
                        const float* __restrict__ x_t, const float* __restrict__ h_f,
                        __bf16* __restrict__ hr_bf, float* __restrict__ u1) {
    const int lane = threadIdx.x & 31;
    const int id = blockIdx.x * 8 + (threadIdx.x >> 5);   // 64 waves: 4 x 16
    const int m0 = (id / 16) * 16, n0 = (id % 16) * 64;
    v8f acc[4] = {};
    accum4(acc, h_bf, U_bf, m0, n0, D_H, lane);
    const int col = lane & 15, rs = (lane >> 4) * 8;
#pragma unroll
    for (int j = 0; j < 4; ++j) {
        int n = n0 + j * 16 + col;
#pragma unroll
        for (int i = 0; i < 8; ++i) {
            int m = m0 + rs + i;
            float v = sigmoidf_(acc[j][i] + x_t[m * D_H2 + n]);
            if (n < D_H) hr_bf[m * D_H + n] = (__bf16)(h_f[m * D_H + n] * v);
            else         u1[m * D_H + (n - D_H)] = v;
        }
    }
}

// ---- step2: h1 = tanh(hr @ Ux^T + xx_t); GRU mix with u1; y-mask ----
__global__ void k_step2(const __bf16* __restrict__ hr_bf, const __bf16* __restrict__ Ux_bf,
                        const float* __restrict__ xx_t, const float* __restrict__ u1,
                        const float* __restrict__ h_f, const float* __restrict__ ym_t,
                        float* __restrict__ h1_f, __bf16* __restrict__ h1_bf) {
    const int lane = threadIdx.x & 31;
    const int id = blockIdx.x * 8 + (threadIdx.x >> 5);   // 32 waves: 4 x 8
    const int m0 = (id / 8) * 16, n0 = (id % 8) * 64;
    v8f acc[4] = {};
    accum4(acc, hr_bf, Ux_bf, m0, n0, D_H, lane);
    const int col = lane & 15, rs = (lane >> 4) * 8;
#pragma unroll
    for (int j = 0; j < 4; ++j) {
        int n = n0 + j * 16 + col;
#pragma unroll
        for (int i = 0; i < 8; ++i) {
            int m = m0 + rs + i;
            float v  = tanhf(acc[j][i] + xx_t[m * D_H + n]);
            float u  = u1[m * D_H + n];
            float hp = h_f[m * D_H + n];
            float h1 = u * hp + (1.f - u) * v;
            float ym = ym_t[m];
            h1 = ym * h1 + (1.f - ym) * hp;
            h1_f[m * D_H + n] = h1;
            h1_bf[m * D_H + n] = (__bf16)h1;
        }
    }
}

// ---- step4: score[tx,b] = x_mask * sum_c tanh(pctx + hq) * U_att ----
// One wave per (tx,b) row; b128 per-lane loads of the L2-resident pctx.
__global__ void k_scores(const __bf16* __restrict__ pctx, const float* __restrict__ hq,
                         const float* __restrict__ uatt, const float* __restrict__ xmask,
                         float* __restrict__ score) {
    const int lane = threadIdx.x & 31;
    const int row  = blockIdx.x * 8 + (threadIdx.x >> 5); // row = tx*B + b
    const int b    = row % N_B;
    const __bf16* p  = pctx + (size_t)row * D_CTX;
    const float* hqb = hq + b * D_CTX;
    float s = 0.f;
    for (int c0 = lane * 8; c0 < D_CTX; c0 += 256) {
        v8bf pv = *(const v8bf*)(p + c0);
        float hv[8], uv[8];
        *(float4*)&hv[0] = *(const float4*)(hqb + c0);
        *(float4*)&hv[4] = *(const float4*)(hqb + c0 + 4);
        *(float4*)&uv[0] = *(const float4*)(uatt + c0);
        *(float4*)&uv[4] = *(const float4*)(uatt + c0 + 4);
#pragma unroll
        for (int i = 0; i < 8; ++i)
            s += tanhf((float)pv[i] + hv[i]) * uv[i];
    }
#pragma unroll
    for (int o = 16; o > 0; o >>= 1) s += __shfl_xor(s, o, 32);
    if (lane == 0) score[row] = s * xmask[row];
}

// ---- step5: softmax over tx per batch column ----
__global__ void k_softmax(const float* __restrict__ score, const float* __restrict__ xmask,
                          float* __restrict__ w) {
    __shared__ float red[256];
    const int b = blockIdx.x, tid = threadIdx.x;
    float s0 = score[tid * N_B + b];
    float s1 = score[(tid + 256) * N_B + b];
    red[tid] = fmaxf(s0, s1); __syncthreads();
    for (int o = 128; o > 0; o >>= 1) { if (tid < o) red[tid] = fmaxf(red[tid], red[tid + o]); __syncthreads(); }
    float mx = red[0]; __syncthreads();
    float e0 = __expf(s0 - mx) * xmask[tid * N_B + b];
    float e1 = __expf(s1 - mx) * xmask[(tid + 256) * N_B + b];
    red[tid] = e0 + e1; __syncthreads();
    for (int o = 128; o > 0; o >>= 1) { if (tid < o) red[tid] += red[tid + o]; __syncthreads(); }
    float inv = 1.f / red[0];
    w[tid * N_B + b] = e0 * inv;
    w[(tid + 256) * N_B + b] = e1 * inv;
}

// ---- step6: atted[b,c] = sum_tx w[tx,b] * context[tx,b,c]; emit atts output ----
// One block per b; each thread owns 4 consecutive c (b64 loads of context).
__global__ void k_attsum(const float* __restrict__ w, const __bf16* __restrict__ ctx_bf,
                         float* __restrict__ atted_f, __bf16* __restrict__ atted_bf,
                         float* __restrict__ atts_out) {
    __shared__ float wl[N_TX];
    const int b = blockIdx.x;
    const int tid = threadIdx.x;
    wl[tid]       = w[tid * N_B + b];
    wl[tid + 256] = w[(tid + 256) * N_B + b];
    __syncthreads();
    const int c0 = tid * 4;
    float a0 = 0.f, a1 = 0.f, a2 = 0.f, a3 = 0.f;
    const __bf16* p = ctx_bf + (size_t)b * D_CTX + c0;
#pragma unroll 4
    for (int tx = 0; tx < N_TX; ++tx) {
        v4bf v = *(const v4bf*)(p + (size_t)tx * N_B * D_CTX);
        float wv = wl[tx];
        a0 += wv * (float)v[0];
        a1 += wv * (float)v[1];
        a2 += wv * (float)v[2];
        a3 += wv * (float)v[3];
    }
    size_t o = (size_t)b * D_CTX + c0;
    *(float4*)(atted_f + o)  = make_float4(a0, a1, a2, a3);
    *(float4*)(atts_out + o) = make_float4(a0, a1, a2, a3);
    v4bf ob; ob[0] = (__bf16)a0; ob[1] = (__bf16)a1; ob[2] = (__bf16)a2; ob[3] = (__bf16)a3;
    *(v4bf*)(atted_bf + o) = ob;
}

// ---- step7: tmp2 = sigmoid(atted@Wc^T + h1@U_nl^T + b_nl); emit h1*r2 (bf16), u2 ----
__global__ void k_step7(const __bf16* __restrict__ atted_bf, const __bf16* __restrict__ Wc_bf,
                        const __bf16* __restrict__ h1_bf, const __bf16* __restrict__ Unl_bf,
                        const float* __restrict__ bnl, const float* __restrict__ h1_f,
                        __bf16* __restrict__ h1r2_bf, float* __restrict__ u2) {
    const int lane = threadIdx.x & 31;
    const int id = blockIdx.x * 8 + (threadIdx.x >> 5);   // 64 waves
    const int m0 = (id / 16) * 16, n0 = (id % 16) * 64;
    v8f acc[4] = {};
    accum4(acc, atted_bf, Wc_bf, m0, n0, D_CTX, lane);
    accum4(acc, h1_bf, Unl_bf, m0, n0, D_H, lane);
    const int col = lane & 15, rs = (lane >> 4) * 8;
#pragma unroll
    for (int j = 0; j < 4; ++j) {
        int n = n0 + j * 16 + col;
#pragma unroll
        for (int i = 0; i < 8; ++i) {
            int m = m0 + rs + i;
            float v = sigmoidf_(acc[j][i] + bnl[n]);
            if (n < D_H) h1r2_bf[m * D_H + n] = (__bf16)(h1_f[m * D_H + n] * v);
            else         u2[m * D_H + (n - D_H)] = v;
        }
    }
}

// ---- step8: h2 = tanh(atted@Wcx^T + (h1*r2)@Ux_nl^T + bx_nl); mix; write outputs ----
__global__ void k_step8(const __bf16* __restrict__ atted_bf, const __bf16* __restrict__ Wcx_bf,
                        const __bf16* __restrict__ h1r2_bf, const __bf16* __restrict__ Uxnl_bf,
                        const float* __restrict__ bxnl, const float* __restrict__ u2,
                        const float* __restrict__ h1_f, const float* __restrict__ ym_t,
                        float* __restrict__ h_f, __bf16* __restrict__ h_bf,
                        float* __restrict__ hs_out, float* __restrict__ ss_out) {
    const int lane = threadIdx.x & 31;
    const int id = blockIdx.x * 8 + (threadIdx.x >> 5);   // 32 waves
    const int m0 = (id / 8) * 16, n0 = (id % 8) * 64;
    v8f acc[4] = {};
    accum4(acc, atted_bf, Wcx_bf, m0, n0, D_CTX, lane);
    accum4(acc, h1r2_bf, Uxnl_bf, m0, n0, D_H, lane);
    const int col = lane & 15, rs = (lane >> 4) * 8;
#pragma unroll
    for (int j = 0; j < 4; ++j) {
        int n = n0 + j * 16 + col;
#pragma unroll
        for (int i = 0; i < 8; ++i) {
            int m = m0 + rs + i;
            float v   = tanhf(acc[j][i] + bxnl[n]);
            float u   = u2[m * D_H + n];
            float h1v = h1_f[m * D_H + n];
            float h2  = u * h1v + (1.f - u) * v;
            float ym  = ym_t[m];
            h2 = ym * h2 + (1.f - ym) * h1v;
            h_f[m * D_H + n]  = h2;
            h_bf[m * D_H + n] = (__bf16)h2;
            hs_out[m * D_H + n] = h2;
            ss_out[m * D_H + n] = h2;
        }
    }
}

extern "C" void kernel_launch(void* const* d_in, const int* in_sizes, int n_in,
                              void* d_out, int out_size, void* d_ws, size_t ws_size,
                              hipStream_t stream) {
    (void)in_sizes; (void)n_in; (void)out_size; (void)ws_size;
    const float* y_emb      = (const float*)d_in[0];
    const float* context    = (const float*)d_in[1];
    const float* init_state = (const float*)d_in[2];
    const float* x_mask     = (const float*)d_in[3];
    const float* y_mask     = (const float*)d_in[4];
    const float* W          = (const float*)d_in[5];
    const float* U          = (const float*)d_in[6];
    const float* b          = (const float*)d_in[7];
    const float* Wx         = (const float*)d_in[8];
    const float* Ux         = (const float*)d_in[9];
    const float* bx         = (const float*)d_in[10];
    const float* Wc_att     = (const float*)d_in[11];
    const float* b_att      = (const float*)d_in[12];
    const float* W_comb_att = (const float*)d_in[13];
    const float* U_att      = (const float*)d_in[14];
    const float* U_nl       = (const float*)d_in[15];
    const float* b_nl       = (const float*)d_in[16];
    const float* Ux_nl      = (const float*)d_in[17];
    const float* bx_nl      = (const float*)d_in[18];
    const float* Wc         = (const float*)d_in[19];
    const float* Wcx        = (const float*)d_in[20];

    float* out  = (float*)d_out;
    float* hs   = out;
    float* ss   = out + (size_t)N_TY * N_B * D_H;
    float* atts = out + (size_t)2 * N_TY * N_B * D_H;

    char* ws = (char*)d_ws;
    size_t off = 0;
    auto alloc = [&](size_t bytes) -> char* {
        char* p = ws + off;
        off = (off + bytes + 255) & ~(size_t)255;
        return p;
    };
    __bf16* ctx_bf   = (__bf16*)alloc((size_t)N_TX * N_B * D_CTX * 2);
    __bf16* pctx_bf  = (__bf16*)alloc((size_t)N_TX * N_B * D_CTX * 2);
    __bf16* yemb_bf  = (__bf16*)alloc((size_t)N_TY * N_B * D_IN * 2);
    __bf16* W_bf     = (__bf16*)alloc((size_t)D_H2 * D_IN * 2);
    __bf16* U_bf     = (__bf16*)alloc((size_t)D_H2 * D_H * 2);
    __bf16* Wx_bf    = (__bf16*)alloc((size_t)D_H * D_IN * 2);
    __bf16* Ux_bf    = (__bf16*)alloc((size_t)D_H * D_H * 2);
    __bf16* Wcatt_bf = (__bf16*)alloc((size_t)D_CTX * D_CTX * 2);
    __bf16* Wcomb_bf = (__bf16*)alloc((size_t)D_CTX * D_H * 2);
    __bf16* Unl_bf   = (__bf16*)alloc((size_t)D_H2 * D_H * 2);
    __bf16* Uxnl_bf  = (__bf16*)alloc((size_t)D_H * D_H * 2);
    __bf16* Wc_bf    = (__bf16*)alloc((size_t)D_H2 * D_CTX * 2);
    __bf16* Wcx_bf   = (__bf16*)alloc((size_t)D_H * D_CTX * 2);
    float*  x_f      = (float*)alloc((size_t)N_TY * N_B * D_H2 * 4);
    float*  xx_f     = (float*)alloc((size_t)N_TY * N_B * D_H * 4);
    float*  h_f      = (float*)alloc((size_t)N_B * D_H * 4);
    __bf16* h_bf     = (__bf16*)alloc((size_t)N_B * D_H * 2);
    float*  h1_f     = (float*)alloc((size_t)N_B * D_H * 4);
    __bf16* h1_bf    = (__bf16*)alloc((size_t)N_B * D_H * 2);
    __bf16* hr_bf    = (__bf16*)alloc((size_t)N_B * D_H * 2);
    float*  u1       = (float*)alloc((size_t)N_B * D_H * 4);
    float*  hq       = (float*)alloc((size_t)N_B * D_CTX * 4);
    float*  score    = (float*)alloc((size_t)N_TX * N_B * 4);
    float*  wbuf     = (float*)alloc((size_t)N_TX * N_B * 4);
    float*  atted_f  = (float*)alloc((size_t)N_B * D_CTX * 4);
    __bf16* atted_bf = (__bf16*)alloc((size_t)N_B * D_CTX * 2);
    float*  u2       = (float*)alloc((size_t)N_B * D_H * 4);
    __bf16* h1r2_bf  = (__bf16*)alloc((size_t)N_B * D_H * 2);

    // --- converts (fp32 -> bf16 resident copies), vectorized x4 ---
    auto cvt = [&](const float* s, __bf16* d, int n) {
        int n4 = n / 4;
        k_cvt<<<(n4 + 255) / 256, 256, 0, stream>>>(s, d, n4);
    };
    cvt(context, ctx_bf, N_TX * N_B * D_CTX);
    cvt(y_emb, yemb_bf, N_TY * N_B * D_IN);
    cvt(W, W_bf, D_H2 * D_IN);
    cvt(U, U_bf, D_H2 * D_H);
    cvt(Wx, Wx_bf, D_H * D_IN);
    cvt(Ux, Ux_bf, D_H * D_H);
    cvt(Wc_att, Wcatt_bf, D_CTX * D_CTX);
    cvt(W_comb_att, Wcomb_bf, D_CTX * D_H);
    cvt(U_nl, Unl_bf, D_H2 * D_H);
    cvt(Ux_nl, Uxnl_bf, D_H * D_H);
    cvt(Wc, Wc_bf, D_H2 * D_CTX);
    cvt(Wcx, Wcx_bf, D_H * D_CTX);

    // --- precompute GEMMs (WMMA) ---
    // pctx = context @ Wc_att^T + b_att  -> bf16 resident
    k_gemm<<<((N_TX * N_B / 16) * (D_CTX / 64)) / 8, 256, 0, stream>>>(
        ctx_bf, Wcatt_bf, b_att, nullptr, pctx_bf, N_TX * N_B, D_CTX, D_CTX);
    // x = y_emb @ W^T + b
    k_gemm<<<((N_TY * N_B / 16) * (D_H2 / 64)) / 8, 256, 0, stream>>>(
        yemb_bf, W_bf, b, x_f, nullptr, N_TY * N_B, D_H2, D_IN);
    // xx = y_emb @ Wx^T + bx
    k_gemm<<<((N_TY * N_B / 16) * (D_H / 64)) / 8, 256, 0, stream>>>(
        yemb_bf, Wx_bf, bx, xx_f, nullptr, N_TY * N_B, D_H, D_IN);

    k_init<<<(N_B * D_H) / 256, 256, 0, stream>>>(init_state, h_f, h_bf);

    // --- recurrence ---
    for (int t = 0; t < N_TY; ++t) {
        const float* x_t  = x_f  + (size_t)t * N_B * D_H2;
        const float* xx_t = xx_f + (size_t)t * N_B * D_H;
        const float* ym_t = y_mask + (size_t)t * N_B;
        float* hs_t   = hs   + (size_t)t * N_B * D_H;
        float* ss_t   = ss   + (size_t)t * N_B * D_H;
        float* atts_t = atts + (size_t)t * N_B * D_CTX;

        k_step1<<<8, 256, 0, stream>>>(h_bf, U_bf, x_t, h_f, hr_bf, u1);
        k_step2<<<4, 256, 0, stream>>>(hr_bf, Ux_bf, xx_t, u1, h_f, ym_t, h1_f, h1_bf);
        // hq = h1 @ W_comb_att^T
        k_gemm<<<8, 256, 0, stream>>>(h1_bf, Wcomb_bf, nullptr, hq, nullptr,
                                      N_B, D_CTX, D_H);
        k_scores<<<(N_TX * N_B) / 8, 256, 0, stream>>>(pctx_bf, hq, U_att, x_mask, score);
        k_softmax<<<N_B, 256, 0, stream>>>(score, x_mask, wbuf);
        k_attsum<<<N_B, 256, 0, stream>>>(wbuf, ctx_bf, atted_f, atted_bf, atts_t);
        k_step7<<<8, 256, 0, stream>>>(atted_bf, Wc_bf, h1_bf, Unl_bf, b_nl, h1_f,
                                       h1r2_bf, u2);
        k_step8<<<4, 256, 0, stream>>>(atted_bf, Wcx_bf, h1r2_bf, Uxnl_bf, bx_nl, u2,
                                       h1_f, ym_t, h_f, h_bf, hs_t, ss_t);
    }
}
